// NonLocal2DBlock_59373627900209
// MI455X (gfx1250) — compile-verified, matrix operand order checked by reading, not run on hardware
//
#include <hip/hip_runtime.h>

typedef _Float16 f16;
typedef __attribute__((ext_vector_type(16))) _Float16 v16h;
typedef __attribute__((ext_vector_type(8)))  _Float16 v8h;
typedef __attribute__((ext_vector_type(8)))  float    v8f;
typedef __attribute__((ext_vector_type(4)))  unsigned int v4u;
typedef __attribute__((ext_vector_type(8)))  int      v8i;
typedef __attribute__((ext_vector_type(4)))  int      v4i;

constexpr int B  = 8;
constexpr int C  = 128;
constexpr int Cp = 16;     // f/g projection channels
constexpr int N  = 4096;   // W*H
constexpr int NCH = 128;   // n-chunk of the h-panel staged in LDS per block

constexpr float LOG2E = 1.44269504088896340736f;

// ---- workspace layout (all offsets multiples of 256 bytes) ----
constexpr size_t OFF_XH = 0;                                  // f16 [B][N][C]   8 MB
constexpr size_t OFF_HH = OFF_XH + (size_t)B * N * C * 2;     // f16 [B][C][N]   8 MB
constexpr size_t OFF_FH = OFF_HH + (size_t)B * C * N * 2;     // f16 [B][N][Cp]  1 MB
constexpr size_t OFF_GH = OFF_FH + (size_t)B * N * Cp * 2;    // f16 [B][N][Cp]  1 MB
constexpr size_t OFF_WH = OFF_GH + (size_t)B * N * Cp * 2;    // f16 [C][C]      32 KB
constexpr size_t OFF_WF = OFF_WH + (size_t)C * C * 2;         // f16 [Cp][C]     4 KB
constexpr size_t OFF_WG = OFF_WF + (size_t)Cp * C * 2;        // f16 [Cp][C]     4 KB
constexpr size_t OFF_L  = OFF_WG + (size_t)Cp * C * 2;        // f32 [B][N]      128 KB

#if defined(__HIP_DEVICE_COMPILE__) && __has_builtin(__builtin_amdgcn_tensor_load_to_lds)
#define HAVE_TDM 1
#else
#define HAVE_TDM 0
#endif

__device__ inline v8f wmma_f16(v16h a, v16h b, v8f c) {
  // D = A(16x32 f16) * B(32x16 f16) + C(16x16 f32)
  return __builtin_amdgcn_wmma_f32_16x16x32_f16(
      /*neg_a=*/false, a, /*neg_b=*/false, b,
      /*c_mod=*/(short)0, c, /*reuse_a=*/false, /*reuse_b=*/false);
}

// Build a K=32 operand from two contiguous 8-half (16B) chunks.
__device__ inline v16h mk16(const f16* lo, const f16* hi) {
  v8h a = *(const v8h*)lo;
  v8h b = *(const v8h*)hi;
  v16h r;
#pragma unroll
  for (int i = 0; i < 8; ++i) { r[i] = a[i]; r[i + 8] = b[i]; }
  return r;
}

// K=16 operand zero-padded to K=32 (upper 8 halves = 0).
__device__ inline v16h mk16lo(const f16* lo) {
  v8h a = *(const v8h*)lo;
  v16h r = {};
#pragma unroll
  for (int i = 0; i < 8; ++i) r[i] = a[i];
  return r;
}

#if HAVE_TDM
// Issue a 2D TDM load: tile (tile1 rows x tile0 elems) of a row-major 2D
// tensor (td1 x td0, row stride stride0 elems), 2-byte elements, into LDS.
// D# bitfields per CDNA5 ISA ch.8 (group0: count/lds_addr/global_addr/type;
// group1: data_size, tensor/tile dims, strides).
__device__ inline void tdm_load_2d(unsigned int lds_off, const void* gsrc,
                                   unsigned int td0, unsigned int td1,
                                   unsigned int tile0, unsigned int tile1,
                                   unsigned long long stride0) {
  unsigned long long ga = (unsigned long long)(uintptr_t)gsrc;
  v4u g0 = {};
  g0[0] = 1u;                                            // count=1, user desc
  g0[1] = lds_off;                                       // lds_addr (bytes)
  g0[2] = (unsigned int)ga;                              // global_addr[31:0]
  g0[3] = (unsigned int)((ga >> 32) & 0x01FFFFFFu) | (2u << 30);  // [56:32]|type=2
  v8i g1 = {};
  g1[0] = (int)(1u << 16);                               // data_size=1 (2 bytes)
  g1[1] = (int)((td0 & 0xFFFFu) << 16);                  // tensor_dim0[15:0]
  g1[2] = (int)(((td0 >> 16) & 0xFFFFu) | ((td1 & 0xFFFFu) << 16));
  g1[3] = (int)(((td1 >> 16) & 0xFFFFu) | ((tile0 & 0xFFFFu) << 16));
  g1[4] = (int)(tile1 & 0xFFFFu);                        // tile_dim1 (tile_dim2=0)
  g1[5] = (int)(unsigned int)stride0;                    // tensor_dim0_stride lo
  g1[6] = (int)(unsigned int)((stride0 >> 32) & 0xFFFFu);
  g1[7] = 0;
  v4i g2 = {}, g3 = {};
#if __clang_major__ >= 23
  v8i g4 = {};
  __builtin_amdgcn_tensor_load_to_lds(g0, g1, g2, g3, g4, 0);
#else
  __builtin_amdgcn_tensor_load_to_lds(g0, g1, g2, g3, 0);
#endif
}
#endif

// ---------------------------------------------------------------------------
// Kernel 1: transpose+convert x [B][C][N] f32 -> xh [B][N][C] f16 (LDS tile)
// ---------------------------------------------------------------------------
__global__ void k_xpose(const float* __restrict__ x, f16* __restrict__ xh) {
  __shared__ float tile[64][65];
  const int b = blockIdx.z, c0 = blockIdx.y * 64, n0 = blockIdx.x * 64;
  for (int i = threadIdx.x; i < 64 * 64; i += blockDim.x) {
    const int ci = i >> 6, ni = i & 63;
    tile[ci][ni] = x[((size_t)(b * C + c0 + ci)) * N + n0 + ni];
  }
  __syncthreads();
  for (int i = threadIdx.x; i < 64 * 64; i += blockDim.x) {
    const int ni = i >> 6, ci = i & 63;
    xh[((size_t)b * N + n0 + ni) * C + c0 + ci] = (f16)tile[ci][ni];
  }
}

// ---------------------------------------------------------------------------
// Kernel 2: convert weights to f16 (row-major [o][c] -> A-operand friendly)
// ---------------------------------------------------------------------------
__global__ void k_wconv(const float* __restrict__ Wf, const float* __restrict__ Wg,
                        const float* __restrict__ Wh,
                        f16* __restrict__ wf16, f16* __restrict__ wg16,
                        f16* __restrict__ wh16) {
  const int tid = blockIdx.x * blockDim.x + threadIdx.x;
  const int stride = gridDim.x * blockDim.x;
  for (int i = tid; i < C * C; i += stride) wh16[i] = (f16)Wh[i];
  for (int i = tid; i < Cp * C; i += stride) {
    wf16[i] = (f16)Wf[i];
    wg16[i] = (f16)Wg[i];
  }
}

// ---------------------------------------------------------------------------
// Kernel 3: WMMA projections. One wave per 16-wide n-tile.
//   f -> [B][N][Cp], PRE-SCALED by log2(e) so both softmax passes use exp2
//   g -> [B][N][Cp] (position-major, WMMA A/B operand layout for scores)
//   h -> [B][C][N]  (row-major along n, WMMA A operand for pass B)
// ---------------------------------------------------------------------------
__global__ void k_proj(const f16* __restrict__ xh, const f16* __restrict__ wf16,
                       const f16* __restrict__ wg16, const f16* __restrict__ wh16,
                       const float* __restrict__ bf, const float* __restrict__ bg,
                       const float* __restrict__ bh,
                       f16* __restrict__ fh, f16* __restrict__ gh, f16* __restrict__ hh) {
  const int b = blockIdx.y;
  const int wv = threadIdx.x >> 5, lane = threadIdx.x & 31;
  const int n_base = (blockIdx.x * 4 + wv) * 16;
  const int row = lane & 15, hi = lane >> 4;

  v8f accF = {}, accG = {};
  v8f accH[8] = {};

  const f16* xrow = xh + ((size_t)b * N + n_base + row) * C;  // B-operand: col = lane%16
#pragma unroll
  for (int k0 = 0; k0 < C; k0 += 32) {
    const v16h bx = mk16(xrow + k0 + hi * 8, xrow + k0 + 16 + hi * 8);
    const v16h af = mk16(wf16 + row * C + k0 + hi * 8, wf16 + row * C + k0 + 16 + hi * 8);
    accF = wmma_f16(af, bx, accF);
    const v16h ag = mk16(wg16 + row * C + k0 + hi * 8, wg16 + row * C + k0 + 16 + hi * 8);
    accG = wmma_f16(ag, bx, accG);
#pragma unroll
    for (int ct = 0; ct < 8; ++ct) {
      const f16* wr = wh16 + (ct * 16 + row) * C;
      const v16h ah = mk16(wr + k0 + hi * 8, wr + k0 + 16 + hi * 8);
      accH[ct] = wmma_f16(ah, bx, accH[ct]);
    }
  }

  // D layout: lane holds (o = i + 8*hi, n = n_base + row)
  const int n = n_base + row;
#pragma unroll
  for (int i = 0; i < 8; ++i) {
    const int o = i + 8 * hi;
    fh[((size_t)b * N + n) * Cp + o] = (f16)((accF[i] + bf[o]) * LOG2E);
    gh[((size_t)b * N + n) * Cp + o] = (f16)(accG[i] + bg[o]);
#pragma unroll
    for (int ct = 0; ct < 8; ++ct) {
      const int c = ct * 16 + o;
      hh[((size_t)b * C + c) * N + n] = (f16)(accH[ct][i] + bh[c]);
    }
  }
}

// ---------------------------------------------------------------------------
// Kernel 4: pass A. Scores arrive in log2 domain (f pre-scaled), so
//   L[n] = max_m s2[n,m] + log2 sum_m exp2(s2 - max).
// One wave per 16-row n-tile; per-lane online (max,sum); one lane-combine.
// ---------------------------------------------------------------------------
__global__ void k_rowstats(const f16* __restrict__ fh, const f16* __restrict__ gh,
                           float* __restrict__ L) {
  const int b = blockIdx.y;
  const int wv = threadIdx.x >> 5, lane = threadIdx.x & 31;
  const int n_base = (blockIdx.x * 8 + wv) * 16;
  const int row = lane & 15, hi = lane >> 4;

  const v16h a = mk16lo(fh + ((size_t)b * N + n_base + row) * Cp + hi * 8);

  float mx[8], sm[8];
#pragma unroll
  for (int i = 0; i < 8; ++i) { mx[i] = -1.0e30f; sm[i] = 0.0f; }

  const f16* gbase = gh + (size_t)b * N * Cp;
  for (int mt = 0; mt < N / 16; ++mt) {
    const v16h bgv = mk16lo(gbase + (size_t)(mt * 16 + row) * Cp + hi * 8);
    v8f z = {};
    const v8f s = wmma_f16(a, bgv, z);
#pragma unroll
    for (int i = 0; i < 8; ++i) {
      const float v = s[i];
      const float nm = fmaxf(mx[i], v);
      sm[i] = sm[i] * exp2f(mx[i] - nm) + exp2f(v - nm);
      mx[i] = nm;
    }
  }
  // combine across the 16 lanes of each half (each holds disjoint m columns)
  for (int off = 1; off < 16; off <<= 1) {
#pragma unroll
    for (int i = 0; i < 8; ++i) {
      const float om = __shfl_xor(mx[i], off, 16);
      const float os = __shfl_xor(sm[i], off, 16);
      const float nm = fmaxf(mx[i], om);
      sm[i] = sm[i] * exp2f(mx[i] - nm) + os * exp2f(om - nm);
      mx[i] = nm;
    }
  }
  if (row == 0) {
#pragma unroll
    for (int i = 0; i < 8; ++i)
      L[(size_t)b * N + n_base + 8 * hi + i] = mx[i] + log2f(sm[i]);
  }
}

// ---------------------------------------------------------------------------
// Kernel 5: pass B + epilogue. One wave per 16-wide m-tile; full C in regs.
//   P[n,m] = exp2(s2[n,m] - L[n]);  attended[:,m] += h[:,n-chunk] x P (WMMA).
// The h-panel chunk (C x NCH, 32 KB) is staged in LDS once per block via the
// Tensor Data Mover (double-buffered, s_wait_tensorcnt), removing the 8x
// per-wave redundancy; all waves consume it with ds_load_b128.
// The score-D layout IS the B-operand layout for the h-GEMM: two consecutive
// 16x16 P tiles pack into the K=32 f16 operand with zero cross-lane traffic.
// ---------------------------------------------------------------------------
__global__ void __launch_bounds__(256)
k_attn(const f16* __restrict__ fh, const f16* __restrict__ gh,
       const f16* __restrict__ hh, const float* __restrict__ L,
       const float* __restrict__ x, const float* __restrict__ gamma,
       float* __restrict__ out) {
  __shared__ float ldsL[N];                                  // 16 KB
  __shared__ __attribute__((aligned(16))) f16 stageH[2][C * NCH];  // 2 x 32 KB
  const int b = blockIdx.y;
  for (int i = threadIdx.x; i < N; i += blockDim.x) ldsL[i] = L[(size_t)b * N + i];

  const int wv = threadIdx.x >> 5, lane = threadIdx.x & 31;
  const int m_base = (blockIdx.x * 8 + wv) * 16;
  const int row = lane & 15, hi = lane >> 4;
  const float gm = gamma[0];

  const f16* fbase = fh + (size_t)b * N * Cp;
  const f16* hbase = hh + (size_t)b * C * N;

  // ---- stage chunk 0 ----
#if HAVE_TDM
  if (wv == 0) {
    tdm_load_2d((unsigned int)(uintptr_t)&stageH[0][0], hbase,
                /*td0=*/N, /*td1=*/C, /*tile0=*/NCH, /*tile1=*/C, /*stride0=*/N);
    __builtin_amdgcn_s_wait_tensorcnt(0);
  }
#else
  for (int t = threadIdx.x; t < C * NCH / 8; t += blockDim.x) {
    const int c = (t * 8) / NCH, nr = (t * 8) % NCH;
    *(v8h*)&stageH[0][c * NCH + nr] = *(const v8h*)&hbase[(size_t)c * N + nr];
  }
#endif
  __syncthreads();

  // fixed B operand for the score GEMM: g columns of this m-tile
  const v16h bgv = mk16lo(gh + ((size_t)b * N + m_base + row) * Cp + hi * 8);

  v8f acc[8] = {};

  for (int ch = 0; ch < N / NCH; ++ch) {
    const int cur = ch & 1;
    // prefetch next chunk into the other buffer
    if (ch + 1 < N / NCH) {
#if HAVE_TDM
      if (wv == 0)
        tdm_load_2d((unsigned int)(uintptr_t)&stageH[cur ^ 1][0],
                    hbase + (size_t)(ch + 1) * NCH,
                    N, C, NCH, C, N);
#else
      for (int t = threadIdx.x; t < C * NCH / 8; t += blockDim.x) {
        const int c = (t * 8) / NCH, nr = (t * 8) % NCH;
        *(v8h*)&stageH[cur ^ 1][c * NCH + nr] =
            *(const v8h*)&hbase[(size_t)c * N + (ch + 1) * NCH + nr];
      }
#endif
    }

    const f16* __restrict__ hs = &stageH[cur][0];
#pragma unroll
    for (int j = 0; j < NCH; j += 32) {
      const int n0 = ch * NCH + j;
      __builtin_prefetch(fbase + (size_t)(n0 + 64 + row) * Cp, 0, 1);
      const v16h a0 = mk16lo(fbase + (size_t)(n0 + row) * Cp + hi * 8);
      const v16h a1 = mk16lo(fbase + (size_t)(n0 + 16 + row) * Cp + hi * 8);
      v8f z = {};
      const v8f s0 = wmma_f16(a0, bgv, z);
      const v8f s1 = wmma_f16(a1, bgv, z);

      v16h pB;
#pragma unroll
      for (int i = 0; i < 8; ++i) {
        pB[i]     = (f16)exp2f(s0[i] - ldsL[n0 + i + 8 * hi]);
        pB[i + 8] = (f16)exp2f(s1[i] - ldsL[n0 + 16 + i + 8 * hi]);
      }
#pragma unroll
      for (int ct = 0; ct < 8; ++ct) {
        const f16* hr = hs + (ct * 16 + row) * NCH + j;
        const v16h ah = mk16(hr + hi * 8, hr + 16 + hi * 8);  // ds_load_b128 x2
        acc[ct] = wmma_f16(ah, pB, acc[ct]);
      }
    }

#if HAVE_TDM
    if (wv == 0) __builtin_amdgcn_s_wait_tensorcnt(0);
#endif
    __syncthreads();
  }

  // epilogue: out = x + gamma * attended
  const int m = m_base + row;
#pragma unroll
  for (int ct = 0; ct < 8; ++ct) {
#pragma unroll
    for (int i = 0; i < 8; ++i) {
      const int c = ct * 16 + i + 8 * hi;
      const size_t idx = ((size_t)b * C + c) * N + m;
      out[idx] = x[idx] + gm * acc[ct][i];
    }
  }
}

// ---------------------------------------------------------------------------
extern "C" void kernel_launch(void* const* d_in, const int* in_sizes, int n_in,
                              void* d_out, int out_size, void* d_ws, size_t ws_size,
                              hipStream_t stream) {
  (void)in_sizes; (void)n_in; (void)out_size; (void)ws_size;
  const float* x     = (const float*)d_in[0];
  const float* Wf    = (const float*)d_in[1];
  const float* bf    = (const float*)d_in[2];
  const float* Wg    = (const float*)d_in[3];
  const float* bg    = (const float*)d_in[4];
  const float* Wh    = (const float*)d_in[5];
  const float* bh    = (const float*)d_in[6];
  const float* gamma = (const float*)d_in[7];

  char* ws = (char*)d_ws;
  f16*   xh   = (f16*)(ws + OFF_XH);
  f16*   hh   = (f16*)(ws + OFF_HH);
  f16*   fhp  = (f16*)(ws + OFF_FH);
  f16*   ghp  = (f16*)(ws + OFF_GH);
  f16*   wh16 = (f16*)(ws + OFF_WH);
  f16*   wf16 = (f16*)(ws + OFF_WF);
  f16*   wg16 = (f16*)(ws + OFF_WG);
  float* Lrow = (float*)(ws + OFF_L);

  k_xpose<<<dim3(N / 64, C / 64, B), 256, 0, stream>>>(x, xh);
  k_wconv<<<dim3(32), 256, 0, stream>>>(Wf, Wg, Wh, wf16, wg16, wh16);
  k_proj<<<dim3(N / 64, B), 128, 0, stream>>>(xh, wf16, wg16, wh16, bf, bg, bh,
                                              fhp, ghp, hh);
  k_rowstats<<<dim3(N / 128, B), 256, 0, stream>>>(fhp, ghp, Lrow);
  k_attn<<<dim3(N / 128, B), 256, 0, stream>>>(fhp, ghp, hh, Lrow, x, gamma,
                                               (float*)d_out);
}
